// Indexer_60129542144614
// MI455X (gfx1250) — matrix operand order repaired and checked by reading, not compiled, not run on hardware
//
#include <hip/hip_runtime.h>
#include <hip/hip_bf16.h>

typedef __attribute__((ext_vector_type(16))) __bf16 v16bf;
typedef __attribute__((ext_vector_type(8)))  float  v8f;
typedef int v4i __attribute__((ext_vector_type(4)));

#define NEG_INF (-__builtin_inff())

// Problem geometry (from reference setup_inputs)
#define B_     2
#define S_     4096
#define D_     2048
#define H_     16
#define HD_    64
#define R_     64
#define RATIO_ 4
#define P_     1024              // S_/RATIO_
#define T_     (B_*S_)           // 8192 tokens
#define KTOP_  256

#if __has_builtin(__builtin_amdgcn_global_load_async_to_lds_b128)
#define HAVE_ASYNC_LDS 1
#else
#define HAVE_ASYNC_LDS 0
#endif

union FragBF {
  uint4 u[2];
  v16bf v;
};

__device__ __forceinline__ unsigned short f2bf(float f) {
  unsigned int u = __float_as_uint(f);
  u += 0x7FFFu + ((u >> 16) & 1u);   // round-to-nearest-even
  return (unsigned short)(u >> 16);
}

// ---------------- converts ----------------
__global__ void cvt_f32_bf16(const float* __restrict__ in,
                             unsigned short* __restrict__ out, int n) {
  int i = blockIdx.x * blockDim.x + threadIdx.x;
  if (i < n) out[i] = f2bf(in[i]);
}

// in [rows, cols] f32 (row-major)  ->  out [cols, rows] bf16 (row-major)
__global__ void cvt_transpose(const float* __restrict__ in,
                              unsigned short* __restrict__ out,
                              int rows, int cols) {
  int i = blockIdx.x * blockDim.x + threadIdx.x;
  if (i < rows * cols) {
    int r = i / cols, c = i - r * cols;
    out[(size_t)c * rows + r] = f2bf(in[i]);
  }
}

// ---------------- generic bf16 WMMA GEMM ----------------
// C[M,N] = A[M,KDIM] * B[KDIM,N], with B given N-major as BT[N,KDIM].
// One wave owns a 16x(16*NGROUP) strip: one A-fragment feeds NGROUP wmmas per K step.
template<int KDIM, int NGROUP, bool BF16OUT>
__global__ void __launch_bounds__(256)
gemm_bf16_wmma(const unsigned short* __restrict__ A,
               const unsigned short* __restrict__ BT,
               void* __restrict__ Cout,
               int Mtiles, int NtileGroups, int ldc) {
  int wid = (int)((blockIdx.x * blockDim.x + threadIdx.x) >> 5);
  if (wid >= Mtiles * NtileGroups) return;       // wave-uniform
  int mt = wid / NtileGroups;
  int ng = wid - mt * NtileGroups;
  int lane = threadIdx.x & 31;
  int ln   = lane & 15;
  int half = lane >> 4;

  const unsigned short* aRow = A + (size_t)(mt * 16 + ln) * KDIM;
  const unsigned short* bRow[NGROUP];
#pragma unroll
  for (int g = 0; g < NGROUP; ++g)
    bRow[g] = BT + (size_t)((ng * NGROUP + g) * 16 + ln) * KDIM;

  v8f acc[NGROUP] = {};
#pragma unroll 2
  for (int k0 = 0; k0 < KDIM; k0 += 32) {
    if (KDIM >= 512) {                          // compile-time
      __builtin_prefetch(aRow + k0 + 512, 0, 0);
      __builtin_prefetch(bRow[0] + k0 + 512, 0, 0);
    }
    FragBF fa;
    // A 16x32 bf16 fragment: two contiguous 8-element runs per lane
    fa.u[0] = *(const uint4*)(aRow + k0 + half * 8);
    fa.u[1] = *(const uint4*)(aRow + k0 + 16 + half * 8);
#pragma unroll
    for (int g = 0; g < NGROUP; ++g) {
      FragBF fb;
      const unsigned short* bp = bRow[g] + k0 + half * 8;
      fb.u[0] = *(const uint4*)(bp);
      fb.u[1] = *(const uint4*)(bp + 16);
      acc[g] = __builtin_amdgcn_wmma_f32_16x16x32_bf16(false, fa.v, false, fb.v,
                                                       (short)0, acc[g], false, false);
    }
  }

#pragma unroll
  for (int g = 0; g < NGROUP; ++g) {
    int ncol = (ng * NGROUP + g) * 16 + ln;
    if (BF16OUT) {
      unsigned short* C = (unsigned short*)Cout;
#pragma unroll
      for (int i = 0; i < 8; ++i) {
        int mrow = mt * 16 + i + half * 8;       // C layout: VGPR i -> M = i + 8*half
        C[(size_t)mrow * ldc + ncol] = f2bf(acc[g][i]);
      }
    } else {
      float* C = (float*)Cout;
#pragma unroll
      for (int i = 0; i < 8; ++i) {
        int mrow = mt * 16 + i + half * 8;
        C[(size_t)mrow * ldc + ncol] = acc[g][i];
      }
    }
  }
}

// ---------------- fused scores + causal mask + top-k ----------------
// One block (8 wave32s) per token. scores[p] = sum_h w_i[h]*relu(q_i[h,:].kcomp[p,:])
__global__ void __launch_bounds__(256)
score_topk_kernel(const unsigned short* __restrict__ qi,    // [T, H*HD] bf16
                  const float* __restrict__ wi,             // [T, H] f32
                  const unsigned short* __restrict__ kcomp, // [B*P, HD] bf16
                  int*  __restrict__ outIdx,                // [T, KTOP]
                  float* __restrict__ outSc) {              // [T, KTOP]
  __shared__ float sc[P_];
  __shared__ int   id[P_];
#if HAVE_ASYNC_LDS
  // per-wave private staging: 16 pools x 64 bf16 = 2KB each
  __shared__ __align__(32) unsigned short kstage[8][16 * HD_];
#endif

  int t    = blockIdx.x;
  int tid  = threadIdx.x;
  int wave = tid >> 5;
  int lane = tid & 31;
  int ln   = lane & 15;
  int half = lane >> 4;

  int b = t / S_;
  int s = t - b * S_;
  int pmax = (s + 1) >> 2;      // pools with window end <= t

  // A fragment: M=16 heads, K=64 Hd (two K=32 fragments)
  const unsigned short* aRow = qi + (size_t)t * (H_ * HD_) + (size_t)ln * HD_;
  FragBF fa0, fa1;
  fa0.u[0] = *(const uint4*)(aRow + half * 8);
  fa0.u[1] = *(const uint4*)(aRow + 16 + half * 8);
  fa1.u[0] = *(const uint4*)(aRow + 32 + half * 8);
  fa1.u[1] = *(const uint4*)(aRow + 48 + half * 8);

  float wv[8];
  {
    const float4* wp = (const float4*)(wi + (size_t)t * H_ + half * 8);
    float4 w0 = wp[0], w1 = wp[1];
    wv[0] = w0.x; wv[1] = w0.y; wv[2] = w0.z; wv[3] = w0.w;
    wv[4] = w1.x; wv[5] = w1.y; wv[6] = w1.z; wv[7] = w1.w;
  }

  const unsigned short* kbase = kcomp + (size_t)b * P_ * HD_;

  for (int chunk = wave; chunk < P_ / 16; chunk += 8) {
    int p0 = chunk * 16;
    if (p0 >= pmax) {                       // wave-uniform: fully masked chunk
      if (lane < 16) sc[p0 + lane] = NEG_INF;
      continue;
    }
    int p = p0 + ln;
#if HAVE_ASYNC_LDS
    // Stage the contiguous 2KB chunk (pools p0..p0+15) into this wave's LDS buffer:
    // 4 x (32 lanes x 16B) async copies, then fence with s_wait_asynccnt.
    {
      const char* src = (const char*)(kbase + (size_t)p0 * HD_);
      char* dst = (char*)&kstage[wave][0];
#pragma unroll
      for (int r2 = 0; r2 < 4; ++r2) {
        int byteOff = (r2 * 32 + lane) * 16;
        __builtin_amdgcn_global_load_async_to_lds_b128(
            (__attribute__((address_space(1))) v4i*)(src + byteOff),
            (__attribute__((address_space(3))) v4i*)(dst + byteOff),
            0, 0);
      }
#if __has_builtin(__builtin_amdgcn_s_wait_asynccnt)
      __builtin_amdgcn_s_wait_asynccnt(0);
#else
      asm volatile("s_wait_asynccnt 0x0" ::: "memory");
#endif
    }
    const unsigned short* kRow = &kstage[wave][(size_t)ln * HD_];
#else
    const unsigned short* kRow = kbase + (size_t)p * HD_;
#endif
    FragBF fb0, fb1;
    fb0.u[0] = *(const uint4*)(kRow + half * 8);
    fb0.u[1] = *(const uint4*)(kRow + 16 + half * 8);
    fb1.u[0] = *(const uint4*)(kRow + 32 + half * 8);
    fb1.u[1] = *(const uint4*)(kRow + 48 + half * 8);

    v8f acc = {};
    acc = __builtin_amdgcn_wmma_f32_16x16x32_bf16(false, fa0.v, false, fb0.v,
                                                  (short)0, acc, false, false);
    acc = __builtin_amdgcn_wmma_f32_16x16x32_bf16(false, fa1.v, false, fb1.v,
                                                  (short)0, acc, false, false);

    float partial = 0.f;
#pragma unroll
    for (int i = 0; i < 8; ++i) partial += wv[i] * fmaxf(acc[i], 0.f);
    float tot = partial + __shfl_xor(partial, 16, 32);   // combine M-halves

    float v = (p < pmax) ? tot : NEG_INF;
    if (lane < 16) sc[p] = v;
  }

  for (int i = tid; i < P_; i += 256) id[i] = i;
  __syncthreads();

  // bitonic sort, descending by score, idx payload
  for (int k = 2; k <= P_; k <<= 1) {
    for (int j = k >> 1; j > 0; j >>= 1) {
      for (int i = tid; i < P_; i += 256) {
        int l = i ^ j;
        if (l > i) {
          bool up = ((i & k) == 0);
          float si = sc[i], sl = sc[l];
          if ((si < sl) == up) {
            sc[i] = sl; sc[l] = si;
            int ti = id[i]; id[i] = id[l]; id[l] = ti;
          }
        }
      }
      __syncthreads();
    }
  }

  // emit top KTOP_ (one element per thread)
  float s0 = sc[tid];
  int idx  = id[tid];
  if (s0 == NEG_INF) idx = -1;
  outIdx[(size_t)t * KTOP_ + tid] = idx;
  outSc [(size_t)t * KTOP_ + tid] = s0;
}

// ---------------- launcher ----------------
extern "C" void kernel_launch(void* const* d_in, const int* in_sizes, int n_in,
                              void* d_out, int out_size, void* d_ws, size_t ws_size,
                              hipStream_t stream) {
  (void)in_sizes; (void)n_in; (void)out_size; (void)ws_size;

  const float* hidden = (const float*)d_in[0];
  const float* w_dq   = (const float*)d_in[1];
  const float* w_iuq  = (const float*)d_in[2];
  const float* w_w    = (const float*)d_in[3];
  const float* w_comp = (const float*)d_in[4];

  char* ws = (char*)d_ws;
  unsigned short* h_bf  = (unsigned short*)(ws);              // 32 MB  [T, D] bf16
  unsigned short* wcT   = (unsigned short*)(ws + 33554432);   // 1 MB   [HD, r*D]
  unsigned short* wdqT  = (unsigned short*)(ws + 34603008);   // 256 KB [R, D]
  unsigned short* wiuqT = (unsigned short*)(ws + 34865152);   // 128 KB [H*HD, R]
  unsigned short* wwT   = (unsigned short*)(ws + 34996224);   // 64 KB  [H, D]
  unsigned short* kcomp = (unsigned short*)(ws + 35061760);   // 256 KB [B*P, HD]
  unsigned short* q_lr  = (unsigned short*)(ws + 35323904);   // 1 MB   [T, R]
  unsigned short* q_i   = (unsigned short*)(ws + 36372480);   // 16 MB  [T, H*HD]
  float*          w_i   = (float*)         (ws + 53149696);   // 512 KB [T, H]

  // 1) converts (hidden to bf16; weights to bf16 N-major for B fragments)
  {
    int n = T_ * D_;
    cvt_f32_bf16<<<dim3((n + 255) / 256), dim3(256), 0, stream>>>(hidden, h_bf, n);
  }
  cvt_transpose<<<dim3((D_ * R_ + 255) / 256), dim3(256), 0, stream>>>(w_dq, wdqT, D_, R_);
  cvt_transpose<<<dim3((R_ * H_ * HD_ + 255) / 256), dim3(256), 0, stream>>>(w_iuq, wiuqT, R_, H_ * HD_);
  cvt_transpose<<<dim3((D_ * H_ + 255) / 256), dim3(256), 0, stream>>>(w_w, wwT, D_, H_);
  cvt_transpose<<<dim3((RATIO_ * D_ * HD_ + 255) / 256), dim3(256), 0, stream>>>(w_comp, wcT, RATIO_ * D_, HD_);

  // 2) kcomp = reshape(hidden,[B*P, r*D]) @ w_comp   -> [B*P, HD] bf16
  {
    int Mt = (B_ * P_) / 16, Ng = 1;          // 128 waves, NGROUP=4 covers N=64
    int waves = Mt * Ng;
    gemm_bf16_wmma<RATIO_ * D_, 4, true><<<dim3((waves * 32 + 255) / 256), dim3(256), 0, stream>>>(
        h_bf, wcT, (void*)kcomp, Mt, Ng, HD_);
  }
  // 3a) q_lr = hidden @ w_dq   -> [T, R] bf16
  {
    int Mt = T_ / 16, Ng = 1;                 // 512 waves, NGROUP=4 covers N=64
    int waves = Mt * Ng;
    gemm_bf16_wmma<D_, 4, true><<<dim3((waves * 32 + 255) / 256), dim3(256), 0, stream>>>(
        h_bf, wdqT, (void*)q_lr, Mt, Ng, R_);
  }
  // 3b) q_i = q_lr @ w_iuq     -> [T, H*HD] bf16
  {
    int Mt = T_ / 16, Ng = (H_ * HD_) / 64;   // 512 x 16 groups of 4 tiles
    int waves = Mt * Ng;
    gemm_bf16_wmma<R_, 4, true><<<dim3((waves * 32 + 255) / 256), dim3(256), 0, stream>>>(
        q_lr, wiuqT, (void*)q_i, Mt, Ng, H_ * HD_);
  }
  // 3c) w_i = hidden @ w_w     -> [T, H] f32
  {
    int Mt = T_ / 16, Ng = 1;                 // 512 waves, single 16-col tile
    int waves = Mt * Ng;
    gemm_bf16_wmma<D_, 1, false><<<dim3((waves * 32 + 255) / 256), dim3(256), 0, stream>>>(
        h_bf, wwT, (void*)w_i, Mt, Ng, H_);
  }

  // 4) fused scores + mask + top-k
  int*   outIdx = (int*)d_out;
  float* outSc  = (float*)d_out + (size_t)T_ * KTOP_;
  score_topk_kernel<<<dim3(T_), dim3(256), 0, stream>>>(q_i, w_i, kcomp, outIdx, outSc);
}
